// STABR_tag_encoder_26697516712557
// MI455X (gfx1250) — matrix operand order repaired and checked by reading, not compiled, not run on hardware
//
#include <hip/hip_runtime.h>
#include <hip/hip_fp16.h>

typedef __attribute__((ext_vector_type(16))) _Float16 v16h;
typedef __attribute__((ext_vector_type(8)))  _Float16 v8h;
typedef __attribute__((ext_vector_type(8)))  float    v8f;

#define B_   256
#define S_   512
#define T_   20
#define E_   25
#define H_   128
#define SKIP_ 64
#define N4H  512   // 4*H
#define KA   224   // combined A K-dim: 32 (x, padded from 25) + 128 (h) + 64 (skip)

// sched_group_barrier masks (LLVM AMDGPU): 0x100 = DS read, 0x008 = MFMA/WMMA
#if __has_builtin(__builtin_amdgcn_sched_group_barrier)
#define SCHED_DS_READ(n)  __builtin_amdgcn_sched_group_barrier(0x100, (n), 0)
#define SCHED_WMMA(n)     __builtin_amdgcn_sched_group_barrier(0x008, (n), 0)
#else
#define SCHED_DS_READ(n)
#define SCHED_WMMA(n)
#endif

__device__ __forceinline__ float sigf(float x) {
    return 1.0f / (1.0f + __expf(-x));
}

// A 16x32 f16 fragment: lanes0-15 K in [0,16); lanes16-31 K in [16,32).
// p already points at this lane's first 8-half chunk.
__device__ __forceinline__ v16h load_afrag(const _Float16* p) {
    v8h lo = *(const v8h*)p;
    v8h hi = *(const v8h*)(p + 8);
    v16h a;
    #pragma unroll
    for (int q = 0; q < 8; ++q) { a[q] = lo[q]; a[8 + q] = hi[q]; }
    return a;
}

// ---------------------------------------------------------------------------
// Kernel 1: embedding gather + masked mean over T tags -> gx (B,S,32) f16
// One wave per (b,s) position, lane = embedding dim.
// ---------------------------------------------------------------------------
__global__ void embed_mean_kernel(const int* __restrict__ tags,
                                  const float* __restrict__ emb,
                                  _Float16* __restrict__ gx) {
    int pos  = blockIdx.x * 8 + (threadIdx.x >> 5);   // (b*S + s)
    int lane = threadIdx.x & 31;
    const int* tp = tags + (size_t)pos * T_;
    float sum = 0.0f;
    int cnt = 0;
    #pragma unroll
    for (int t = 0; t < T_; ++t) {
        int tag = tp[t];
        if (tag != 0) {
            ++cnt;
            if (lane < E_) sum += emb[(size_t)tag * E_ + lane];
        }
    }
    float denom = (cnt == 0) ? 1.0f : (float)cnt;
    _Float16 v = (lane < E_) ? (_Float16)(sum / denom) : (_Float16)0.0f;
    gx[(size_t)pos * 32 + lane] = v;
}

// ---------------------------------------------------------------------------
// Kernel 2: f32 -> f16 convert for embedded_skips -> gs (B,S,64) f16
// ---------------------------------------------------------------------------
__global__ void skips_to_f16_kernel(const float* __restrict__ src,
                                    _Float16* __restrict__ dst) {
    int i = blockIdx.x * blockDim.x + threadIdx.x;   // one float4 per thread
    float4 v = ((const float4*)src)[i];
    float tmp[4] = {v.x, v.y, v.z, v.w};
    #pragma unroll
    for (int j = 0; j < 4; ++j) dst[(size_t)i * 4 + j] = (_Float16)tmp[j];
}

// ---------------------------------------------------------------------------
// Kernel 3: LSTM (both directions via blockIdx.y).
// Block = 256 threads (8 waves), owns a 16-row batch tile for all 512 steps.
// Wave w owns H-columns [16w,16w+16): computes the i/f/g/o gate tiles for
// those columns each step via v_wmma_f32_16x16x32_f16.
// LDS: WB[512][224] f16 weights (transposed -> B-frag = contiguous 32B read),
//      AT[16][224] f16 combined A (x|h|skip), mask[16] f32.
// k-loop: double-buffered + sched_group_barrier-pinned pipeline; first tier
// uses inline-zero C; bias added in the elementwise phase.
// __launch_bounds__(256,1): 236KB LDS means exactly 1 block/WGP can ever be
// resident, so let the allocator spend VGPRs freely (kill residual spills).
// ---------------------------------------------------------------------------
__global__ void
__launch_bounds__(256, 1)
lstm_kernel(const _Float16* __restrict__ gx,
            const _Float16* __restrict__ gs,
            const int* __restrict__ tags,
            const float* __restrict__ Kf, const float* __restrict__ Rf,
            const float* __restrict__ Sf, const float* __restrict__ bf_,
            const float* __restrict__ Kb, const float* __restrict__ Rb,
            const float* __restrict__ Sb, const float* __restrict__ bb_,
            float* __restrict__ out_buf,   // (B,S,256)
            float* __restrict__ h_fwd) {   // (B,128)
    extern __shared__ _Float16 smL[];
    _Float16* WB = smL;                    // 512*224 halves
    _Float16* AT = smL + (size_t)N4H * KA; // 16*224 halves
    float* maskL = (float*)(AT + 16 * KA); // 16 floats

    const int tid  = threadIdx.x;
    const int wave = tid >> 5;
    const int lane = tid & 31;
    const int nloc = lane & 15;       // N within a 16x16 tile
    const int khalf = lane >> 4;      // lane-half selector for frag layouts
    const int b0  = blockIdx.x * 16;
    const int dir = blockIdx.y;

    const float* Kw = dir ? Kb : Kf;
    const float* Rw = dir ? Rb : Rf;
    const float* Sw = dir ? Sb : Sf;
    const float* bw = dir ? bb_ : bf_;

    // ---- stage transposed f16 weights into LDS: WB[c][k], k = [x32|h128|sk64]
    for (int idx = tid; idx < N4H * KA; idx += 256) {
        int c = idx / KA, k = idx - c * KA;
        float v;
        if (k < 32)        v = (k < E_) ? Kw[(size_t)k * N4H + c] : 0.0f;
        else if (k < 160)  v = Rw[(size_t)(k - 32) * N4H + c];
        else               v = Sw[(size_t)(k - 160) * N4H + c];
        WB[(size_t)c * KA + k] = (_Float16)v;
    }
    // zero A tile (h section starts at 0)
    for (int idx = tid; idx < 16 * KA; idx += 256) AT[idx] = (_Float16)0.0f;
    __syncthreads();

    // ---- per-gate LDS base pointers for B-fragments (loop-invariant address)
    // B 32x16 layout: lane%16 = N col; halves = 16 contiguous K per half-wave.
    const _Float16* bp0 = WB + (size_t)(0 * H_ + wave * 16 + nloc) * KA + khalf * 16;
    const _Float16* bp1 = WB + (size_t)(1 * H_ + wave * 16 + nloc) * KA + khalf * 16;
    const _Float16* bp2 = WB + (size_t)(2 * H_ + wave * 16 + nloc) * KA + khalf * 16;
    const _Float16* bp3 = WB + (size_t)(3 * H_ + wave * 16 + nloc) * KA + khalf * 16;
    const _Float16* aptr = AT + (size_t)nloc * KA + khalf * 16;

    float bv[4];
    #pragma unroll
    for (int g = 0; g < 4; ++g) bv[g] = bw[g * H_ + wave * 16 + nloc];

    float cst[8], hst[8];
    #pragma unroll
    for (int r = 0; r < 8; ++r) { cst[r] = 0.0f; hst[r] = 0.0f; }

    for (int i = 0; i < S_; ++i) {
        int s = dir ? (S_ - 1 - i) : i;
        // ---- cooperative staging of x (16x32) and skip (16x64) tiles + mask
        {
            int row = tid >> 4, d = tid & 15;
            const unsigned* src = (const unsigned*)(gx + ((size_t)(b0 + row) * S_ + s) * 32);
            ((unsigned*)(AT + (size_t)row * KA))[d] = src[d];
        }
        #pragma unroll
        for (int j = 0; j < 2; ++j) {
            int idx = tid + j * 256;
            int row = idx >> 5, d = idx & 31;
            const unsigned* src = (const unsigned*)(gs + ((size_t)(b0 + row) * S_ + s) * 64);
            ((unsigned*)(AT + (size_t)row * KA + 160))[d] = src[d];
        }
        if (tid < 16)
            maskL[tid] = (tags[((size_t)(b0 + tid) * S_ + s) * T_] != 0) ? 1.0f : 0.0f;
        __syncthreads();

        // ---- z = [x|h|skip] @ WB : 7 k-steps x 4 gate tiles, pipelined.
        // First tier uses C = inline 0 (no init movs); bias added later.
        v8f acc[4];
        v16h a_cur = load_afrag(aptr);
        v16h b0c = *(const v16h*)bp0;
        v16h b1c = *(const v16h*)bp1;
        v16h b2c = *(const v16h*)bp2;
        v16h b3c = *(const v16h*)bp3;

        #pragma unroll
        for (int t = 0; t < 7; ++t) {
            v16h a_nxt, b0n, b1n, b2n, b3n;
            if (t < 6) {
                int o = (t + 1) * 32;
                a_nxt = load_afrag(aptr + o);
                b0n = *(const v16h*)(bp0 + o);
                b1n = *(const v16h*)(bp1 + o);
                b2n = *(const v16h*)(bp2 + o);
                b3n = *(const v16h*)(bp3 + o);
            }
            if (t == 0) {
                v8f z = {};
                acc[0] = __builtin_amdgcn_wmma_f32_16x16x32_f16(
                    false, a_cur, false, b0c, (short)0, z, false, false);
                acc[1] = __builtin_amdgcn_wmma_f32_16x16x32_f16(
                    false, a_cur, false, b1c, (short)0, z, false, false);
                acc[2] = __builtin_amdgcn_wmma_f32_16x16x32_f16(
                    false, a_cur, false, b2c, (short)0, z, false, false);
                acc[3] = __builtin_amdgcn_wmma_f32_16x16x32_f16(
                    false, a_cur, false, b3c, (short)0, z, false, false);
            } else {
                acc[0] = __builtin_amdgcn_wmma_f32_16x16x32_f16(
                    false, a_cur, false, b0c, (short)0, acc[0], false, false);
                acc[1] = __builtin_amdgcn_wmma_f32_16x16x32_f16(
                    false, a_cur, false, b1c, (short)0, acc[1], false, false);
                acc[2] = __builtin_amdgcn_wmma_f32_16x16x32_f16(
                    false, a_cur, false, b2c, (short)0, acc[2], false, false);
                acc[3] = __builtin_amdgcn_wmma_f32_16x16x32_f16(
                    false, a_cur, false, b3c, (short)0, acc[3], false, false);
            }
            a_cur = a_nxt; b0c = b0n; b1c = b1n; b2c = b2n; b3c = b3n;
        }
        // pin pipeline: DS(t0) | DS(t1) WMMA(t0) | ... | DS(t6) WMMA(t5) | WMMA(t6)
        SCHED_DS_READ(10);
        #pragma unroll
        for (int t = 0; t < 6; ++t) {
            SCHED_DS_READ(10);
            SCHED_WMMA(4);
        }
        SCHED_WMMA(4);
        __syncthreads();   // all A-frag reads done before h rewrite

        // ---- gates + state update; C/D tile: row M = r + 8*khalf, col = nloc
        #pragma unroll
        for (int r = 0; r < 8; ++r) {
            int M = r + khalf * 8;
            float zi = acc[0][r] + bv[0];
            float zf = acc[1][r] + bv[1];
            float zg = acc[2][r] + bv[2];
            float zo = acc[3][r] + bv[3];
            float cn = sigf(zf) * cst[r] + sigf(zi) * tanhf(zg);
            float hn = sigf(zo) * tanhf(cn);
            float m  = maskL[M];
            float c2 = m * cn + (1.0f - m) * cst[r];
            float h2 = m * hn + (1.0f - m) * hst[r];
            cst[r] = c2; hst[r] = h2;
            int hc = wave * 16 + nloc;
            AT[(size_t)M * KA + 32 + hc] = (_Float16)h2;
            out_buf[(((size_t)(b0 + M)) * S_ + s) * 256 + dir * H_ + hc] = h2;
        }
        __syncthreads();   // h/mask writes visible before next staging
    }

    if (dir == 0) {
        #pragma unroll
        for (int r = 0; r < 8; ++r) {
            int M = r + khalf * 8;
            h_fwd[(size_t)(b0 + M) * H_ + wave * 16 + nloc] = hst[r];
        }
    }
}

// ---------------------------------------------------------------------------
// Kernel 4: attention. One block (8 waves) per batch row.
// keys = out @ Wk via WMMA (M=16 s-rows, K=256, N=128 over 8 waves),
// energy = tanh(keys+q)@We, masked softmax over S, context = sum w*out.
// bk is folded into the q term (tanh(key + q + bk)), so acc starts at zero.
// ---------------------------------------------------------------------------
__global__ void
__launch_bounds__(256, 2)
attn_kernel(const float* __restrict__ out_buf,
            const float* __restrict__ h_fwd,
            const int* __restrict__ tags,
            const float* __restrict__ Wk, const float* __restrict__ bk,
            const float* __restrict__ Wq, const float* __restrict__ bq,
            const float* __restrict__ We, const float* __restrict__ be,
            float* __restrict__ ctx_out) {
    extern __shared__ _Float16 smA[];
    _Float16* WkT  = smA;                    // 128*256 halves (transposed Wk)
    _Float16* AT2  = smA + 128 * 256;        // 16*256 halves
    float* q       = (float*)(AT2 + 16 * 256);   // 128
    float* e_all   = q + 128;                    // 512
    float* red     = e_all + 512;                // 16

    const int b    = blockIdx.x;
    const int tid  = threadIdx.x;
    const int wave = tid >> 5;
    const int lane = tid & 31;
    const int nloc = lane & 15;
    const int khalf = lane >> 4;

    for (int idx = tid; idx < 128 * 256; idx += 256) {
        int n = idx >> 8, k = idx & 255;
        WkT[(size_t)n * 256 + k] = (_Float16)Wk[(size_t)k * 128 + n];
    }
    if (tid < 128) {
        float a = bq[tid];
        for (int k = 0; k < 128; ++k) a += h_fwd[(size_t)b * 128 + k] * Wq[(size_t)k * 128 + tid];
        q[tid] = a;
    }
    for (int i = tid; i < 512; i += 256) {
        float m = (tags[((size_t)b * S_ + i) * T_] != 0) ? 0.0f : -1e9f;
        e_all[i] = be[0] + m;
    }
    __syncthreads();

    float wev = We[wave * 16 + nloc];
    float qv  = q[wave * 16 + nloc] + bk[wave * 16 + nloc];   // fold bk into q
    const _Float16* aptr2 = AT2 + (size_t)nloc * 256 + khalf * 16;
    const _Float16* bptr2 = WkT + (size_t)(wave * 16 + nloc) * 256 + khalf * 16;

    for (int sc = 0; sc < 32; ++sc) {
        {   // stage out-tile (16 s-rows x 256 features) as f16
            int row = tid >> 4, c0 = (tid & 15) * 16;
            const float* src = out_buf + ((size_t)b * S_ + sc * 16 + row) * 256 + c0;
            _Float16* dst = AT2 + (size_t)row * 256 + c0;
            #pragma unroll
            for (int j = 0; j < 16; ++j) dst[j] = (_Float16)src[j];
        }
        __syncthreads();

        v8f acc;
        v16h a_cur = load_afrag(aptr2);
        v16h b_cur = *(const v16h*)bptr2;
        #pragma unroll
        for (int t = 0; t < 8; ++t) {
            v16h a_nxt, b_nxt;
            if (t < 7) {
                int o = (t + 1) * 32;
                a_nxt = load_afrag(aptr2 + o);
                b_nxt = *(const v16h*)(bptr2 + o);
            }
            if (t == 0) {
                v8f z = {};
                acc = __builtin_amdgcn_wmma_f32_16x16x32_f16(
                    false, a_cur, false, b_cur, (short)0, z, false, false);
            } else {
                acc = __builtin_amdgcn_wmma_f32_16x16x32_f16(
                    false, a_cur, false, b_cur, (short)0, acc, false, false);
            }
            a_cur = a_nxt; b_cur = b_nxt;
        }
        SCHED_DS_READ(4);
        #pragma unroll
        for (int t = 0; t < 7; ++t) {
            SCHED_DS_READ(4);
            SCHED_WMMA(1);
        }
        SCHED_WMMA(1);

        #pragma unroll
        for (int r = 0; r < 8; ++r) {
            float v = tanhf(acc[r] + qv) * wev;
            v += __shfl_xor(v, 1, 16);
            v += __shfl_xor(v, 2, 16);
            v += __shfl_xor(v, 4, 16);
            v += __shfl_xor(v, 8, 16);
            if (nloc == 0) atomicAdd(&e_all[sc * 16 + r + khalf * 8], v);
        }
        __syncthreads();
    }

    // masked softmax over S in LDS
    float mx = fmaxf(e_all[tid], e_all[tid + 256]);
    #pragma unroll
    for (int off = 16; off; off >>= 1) mx = fmaxf(mx, __shfl_xor(mx, off, 32));
    if (lane == 0) red[wave] = mx;
    __syncthreads();
    if (tid == 0) {
        float m2 = red[0];
        for (int w2 = 1; w2 < 8; ++w2) m2 = fmaxf(m2, red[w2]);
        red[8] = m2;
    }
    __syncthreads();
    float Mv = red[8];
    float sv = __expf(e_all[tid] - Mv) + __expf(e_all[tid + 256] - Mv);
    #pragma unroll
    for (int off = 16; off; off >>= 1) sv += __shfl_xor(sv, off, 32);
    if (lane == 0) red[wave] = sv;
    __syncthreads();
    if (tid == 0) {
        float s2 = 0.0f;
        for (int w2 = 0; w2 < 8; ++w2) s2 += red[w2];
        red[9] = s2;
    }
    __syncthreads();
    float inv = 1.0f / red[9];
    e_all[tid]       = __expf(e_all[tid] - Mv) * inv;
    e_all[tid + 256] = __expf(e_all[tid + 256] - Mv) * inv;
    __syncthreads();

    // context[j] = sum_s w[s] * out[b,s,j]; thread = feature j (coalesced)
    float ctx = 0.0f;
    for (int s = 0; s < S_; ++s)
        ctx += e_all[s] * out_buf[((size_t)b * S_ + s) * 256 + tid];
    ctx_out[(size_t)b * 256 + tid] = ctx;
}

// ---------------------------------------------------------------------------
extern "C" void kernel_launch(void* const* d_in, const int* in_sizes, int n_in,
                              void* d_out, int out_size, void* d_ws, size_t ws_size,
                              hipStream_t stream) {
    const int*   tags  = (const int*)  d_in[0];
    const float* skips = (const float*)d_in[1];
    const float* emb   = (const float*)d_in[2];
    const float* Kf    = (const float*)d_in[3];
    const float* Rf    = (const float*)d_in[4];
    const float* Sf    = (const float*)d_in[5];
    const float* bfv   = (const float*)d_in[6];
    const float* Kb    = (const float*)d_in[7];
    const float* Rb    = (const float*)d_in[8];
    const float* Sb    = (const float*)d_in[9];
    const float* bbv   = (const float*)d_in[10];
    const float* Wk    = (const float*)d_in[11];
    const float* bk    = (const float*)d_in[12];
    const float* Wq    = (const float*)d_in[13];
    const float* bq    = (const float*)d_in[14];
    const float* We    = (const float*)d_in[15];
    const float* be    = (const float*)d_in[16];
    float* ctx_out = (float*)d_out;

    char* w = (char*)d_ws;
    _Float16* gx = (_Float16*)w;          w += (size_t)B_ * S_ * 32 * sizeof(_Float16);
    _Float16* gs = (_Float16*)w;          w += (size_t)B_ * S_ * 64 * sizeof(_Float16);
    float* out_buf = (float*)w;           w += (size_t)B_ * S_ * 256 * sizeof(float);
    float* h_fwd = (float*)w;             w += (size_t)B_ * 128 * sizeof(float);

    embed_mean_kernel<<<(B_ * S_) / 8, 256, 0, stream>>>(tags, emb, gx);
    skips_to_f16_kernel<<<(B_ * S_ * 64 / 4) / 256, 256, 0, stream>>>(skips, gs);

    size_t lds_lstm = (size_t)N4H * KA * 2 + 16 * KA * 2 + 16 * 4;   // 236,608 B < 320 KB
    lstm_kernel<<<dim3(16, 2), 256, lds_lstm, stream>>>(
        gx, gs, tags, Kf, Rf, Sf, bfv, Kb, Rb, Sb, bbv, out_buf, h_fwd);

    size_t lds_attn = (size_t)128 * 256 * 2 + 16 * 256 * 2 + (128 + 512 + 16) * 4;
    attn_kernel<<<B_, 256, lds_attn, stream>>>(
        out_buf, h_fwd, tags, Wk, bk, Wq, bq, We, be, ctx_out);
}
